// FaithfulAttentionPooling_46935402611184
// MI455X (gfx1250) — matrix operand order repaired and checked by reading, not compiled, not run on hardware
//
#include <hip/hip_runtime.h>
#include <hip/hip_bf16.h>

// ---------------- problem constants ----------------
#define Bdim 16
#define Sdim 2048
#define Hdim 768
#define Ddim 50
#define Dpad 64            // padded D for WMMA (2 x K=32 steps)
#define BIGC 999999.0f
#define PFRAC 0.2f
#define DROPP 0.2f
#define INV_KEEP 1.25f     // 1/(1-DROP)
#define SCALE 0.14142135623730950488f  // 1/sqrt(50)
#define LDSTH 776          // padded f16 LDS row stride: lane m -> dword bank 4m%64,
                           // 16 lanes cover disjoint 4-bank groups (conflict-free b128)

typedef __attribute__((ext_vector_type(16))) _Float16 v16h;
typedef __attribute__((ext_vector_type(8)))  _Float16 v8h;
typedef __attribute__((ext_vector_type(4)))  _Float16 v4h;
typedef __attribute__((ext_vector_type(8)))  float    v8f;
typedef __attribute__((ext_vector_type(4)))  float    v4f;

// deterministic 32-bit hash dropout; (b,q,k) index fits in 26 bits
__device__ __forceinline__ float rand_u01(unsigned x) {
    x += 0x9E3779B9u * 42u;
    x ^= x >> 16; x *= 0x7FEB352Du;
    x ^= x >> 15; x *= 0x846CA68Bu;
    x ^= x >> 16;
    return (float)(x >> 8) * (1.0f / 16777216.0f);
}

// ---------------- kernel 0: build padded/transposed f16 weights -------------
// Wt[128][768]: rows 0..63 = Wq columns (0 for col>=50), rows 64..127 = Wk cols
__global__ __launch_bounds__(256)
void wt_kernel(const float* __restrict__ Wq, const float* __restrict__ Wk,
               _Float16* __restrict__ Wt) {
    const int n = blockIdx.x;          // 0..127
    const int col = n & 63;
    const float* W = (n < 64) ? Wq : Wk;
    for (int h = threadIdx.x; h < Hdim; h += 256) {
        float v = (col < Ddim) ? W[h * Ddim + col] : 0.0f;
        Wt[(size_t)n * Hdim + h] = (_Float16)v;
    }
}

// ---------------- kernel 1: WMMA Q/K projection ----------------------------
// grid (B*S/16); block 256 = 8 waves, one N-tile of 16 output cols per wave.
// 16 vec rows converted once to f16 at LDS-fill; 24 accumulating K-steps.
__global__ __launch_bounds__(256)
void proj_kernel(const float* __restrict__ vec, const _Float16* __restrict__ Wt,
                 const float* __restrict__ bq, const float* __restrict__ bk,
                 _Float16* __restrict__ qf, _Float16* __restrict__ kf) {
    const int row0 = blockIdx.x * 16;  // base row in flattened [B*S]
    __shared__ _Float16 ldsH[16 * LDSTH];

    // fill: f32 global -> f16 LDS, converted exactly once (vectorized x4)
    for (int i = threadIdx.x * 4; i < 16 * Hdim; i += 1024) {
        int r = i / Hdim, c0 = i - r * Hdim;           // quad stays inside one row
        v4f x = *(const v4f*)(vec + (size_t)row0 * Hdim + i);
        v4h y;
#pragma unroll
        for (int e = 0; e < 4; ++e) y[e] = (_Float16)x[e];
        *(v4h*)(&ldsH[r * LDSTH + c0]) = y;
    }
    __syncthreads();

    const int lane = threadIdx.x & 31;
    const int nt = threadIdx.x >> 5;   // wave id == N-tile id (0..7)
    const int nlo = lane & 15;
    const int hiHalf = lane >> 4;
    const int colG = nt * 16 + nlo;    // 0..63 -> q, 64..127 -> k

    const _Float16* lr = &ldsH[nlo * LDSTH + hiHalf * 16];         // A rows
    const _Float16* wrow = Wt + (size_t)colG * Hdim + hiHalf * 16; // B rows

    v8f c = {};
#pragma unroll 4
    for (int ks = 0; ks < Hdim / 32; ++ks) {
        const _Float16* p = lr + ks * 32;
        v8h y0 = *(const v8h*)(p);          // K = kbase .. kbase+7
        v8h y1 = *(const v8h*)(p + 8);      // K = kbase+8 .. kbase+15
        v16h a;
#pragma unroll
        for (int i = 0; i < 8; ++i) { a[i] = y0[i]; a[8 + i] = y1[i]; }
        v16h bfrag = *(const v16h*)(wrow + ks * 32);
        c = __builtin_amdgcn_wmma_f32_16x16x32_f16(false, a, false, bfrag,
                                                   (short)0, c, false, false);
    }

    const int col = colG & 63;
    const bool isK = colG >= 64;
    float bias = (col < Ddim) ? (isK ? bk[col] : bq[col]) : 0.0f;
    _Float16* outp = (isK ? kf : qf) + (size_t)row0 * Dpad + col;
#pragma unroll
    for (int j = 0; j < 8; ++j) {
        int M = j + 8 * hiHalf;
        outp[(size_t)M * Dpad] = (_Float16)(c[j] + bias);
    }
}

// ---------------- kernel 2: WMMA scores + per-lane online softmax ----------
// grid (S/16, B), block 256 (8 waves). Each wave: 16 key tiles, double-buffered
// B fragments; per-lane (m,Z,W) state, single 16-lane merge at the end.
__global__ __launch_bounds__(256)
void score_kernel(const _Float16* __restrict__ qf, const _Float16* __restrict__ kf,
                  const int* __restrict__ mask, float* __restrict__ ts_raw) {
    const int q0 = blockIdx.x * 16;
    const int b = blockIdx.y;
    const int lane = threadIdx.x & 31;
    const int wave = threadIdx.x >> 5;
    const int nlo = lane & 15;
    const int hiHalf = lane >> 4;

    // A fragments: q rows (contiguous k = 16*hiHalf + s, same order as B)
    const _Float16* qrow = qf + ((size_t)b * Sdim + q0 + nlo) * Dpad;
    const int kbase = hiHalf * 16;
    v8h lo1 = *(const v8h*)(qrow + kbase);
    v8h hi1 = *(const v8h*)(qrow + kbase + 8);
    v8h lo2 = *(const v8h*)(qrow + kbase + 32);
    v8h hi2 = *(const v8h*)(qrow + kbase + 40);
    v16h a1, a2;
#pragma unroll
    for (int i = 0; i < 8; ++i) {
        a1[i] = lo1[i]; a1[i + 8] = hi1[i];
        a2[i] = lo2[i]; a2[i + 8] = hi2[i];
    }

    float mrun[8], Zr[8], Wr[8];
#pragma unroll
    for (int j = 0; j < 8; ++j) { mrun[j] = -1.0e30f; Zr[j] = 0.0f; Wr[j] = 0.0f; }

    const int koff = hiHalf * 16;
    const int nkt = Sdim / 16;

    // prefetch first tile
    int kt = wave;
    v16h b1c, b2c; float madd_c;
    {
        const int key = kt * 16 + nlo;
        const _Float16* krow = kf + ((size_t)b * Sdim + key) * Dpad;
        b1c = *(const v16h*)(krow + koff);
        b2c = *(const v16h*)(krow + koff + 32);
        madd_c = BIGC * ((float)mask[b * Sdim + key] - 1.0f);
    }

    for (; kt < nkt; kt += 8) {
        // prefetch next tile (clamped; redundant reload on last iteration)
        const int ktn = (kt + 8 < nkt) ? kt + 8 : kt;
        v16h b1n, b2n; float madd_n;
        {
            const int key = ktn * 16 + nlo;
            const _Float16* krow = kf + ((size_t)b * Sdim + key) * Dpad;
            b1n = *(const v16h*)(krow + koff);
            b2n = *(const v16h*)(krow + koff + 32);
            madd_n = BIGC * ((float)mask[b * Sdim + key] - 1.0f);
        }

        v8f c = {};
        c = __builtin_amdgcn_wmma_f32_16x16x32_f16(false, a1, false, b1c,
                                                   (short)0, c, false, false);
        c = __builtin_amdgcn_wmma_f32_16x16x32_f16(false, a2, false, b2c,
                                                   (short)0, c, false, false);

        // 26-bit unique (b,q,k) index: (b*S + q) << 11 | key
        const unsigned linbase =
            ((unsigned)(b * Sdim + q0 + 8 * hiHalf) << 11) | (unsigned)(kt * 16 + nlo);
#pragma unroll
        for (int j = 0; j < 8; ++j) {
            float s = c[j] * SCALE + madd_c;
            float keep = (rand_u01(linbase + ((unsigned)j << 11)) < (1.0f - DROPP))
                             ? 1.0f : 0.0f;
            float mo = mrun[j];
            float mn = fmaxf(mo, s);
            float e = __expf(s - mn);
            float sc = __expf(mo - mn);
            Zr[j] = Zr[j] * sc + e;
            Wr[j] = Wr[j] * sc + e * keep;
            mrun[j] = mn;
        }
        b1c = b1n; b2c = b2n; madd_c = madd_n;
    }

    // one-time 16-lane butterfly merge of (m,Z,W) per row
#pragma unroll
    for (int j = 0; j < 8; ++j) {
        float m = mrun[j], Z = Zr[j], W = Wr[j];
#pragma unroll
        for (int d = 1; d < 16; d <<= 1) {
            float mo = __shfl_xor(m, d, 32);
            float zo = __shfl_xor(Z, d, 32);
            float wo = __shfl_xor(W, d, 32);
            float mn = fmaxf(m, mo);
            float s0 = __expf(m - mn), s1 = __expf(mo - mn);
            Z = Z * s0 + zo * s1;
            W = W * s0 + wo * s1;
            m = mn;
        }
        mrun[j] = m; Zr[j] = Z; Wr[j] = W;
    }

    __shared__ float red[8][16][3];
    if (nlo == 0) {
#pragma unroll
        for (int j = 0; j < 8; ++j) {
            int r = j + 8 * hiHalf;
            red[wave][r][0] = mrun[j];
            red[wave][r][1] = Zr[j];
            red[wave][r][2] = Wr[j];
        }
    }
    __syncthreads();
    if (threadIdx.x < 16) {
        const int r = threadIdx.x;
        float M = -1.0e30f, Z = 0.0f, W = 0.0f;
#pragma unroll
        for (int w = 0; w < 8; ++w) {
            float mw = red[w][r][0], zw = red[w][r][1], ww = red[w][r][2];
            float mn = fmaxf(M, mw);
            float s0 = __expf(M - mn), s1 = __expf(mw - mn);
            Z = Z * s0 + zw * s1;
            W = W * s0 + ww * s1;
            M = mn;
        }
        ts_raw[(size_t)b * Sdim + q0 + r] = (W / Z) * INV_KEEP;
    }
}

// ---------------- kernel 3: per-batch sort / threshold / normalize ----------
__global__ __launch_bounds__(256)
void select_kernel(const float* __restrict__ ts_raw, const int* __restrict__ mask,
                   float* __restrict__ ts_out, float* __restrict__ expl,
                   float* __restrict__ stats) {
    const int b = blockIdx.x;
    const int t = threadIdx.x;
    __shared__ float v[Sdim];
    __shared__ float sred[256];
    __shared__ int ired[256];

    float tsum = 0.0f; int msum = 0;
    for (int i = t; i < Sdim; i += 256) {
        float x = ts_raw[(size_t)b * Sdim + i];
        v[i] = x; tsum += x;
        msum += mask[b * Sdim + i];
    }
    sred[t] = tsum; ired[t] = msum;
    __syncthreads();
    for (int o = 128; o > 0; o >>= 1) {
        if (t < o) { sred[t] += sred[t + o]; ired[t] += ired[t + o]; }
        __syncthreads();
    }
    const float T = sred[0];
    const int ntok = ired[0];
    const int kper = (int)(PFRAC * (float)ntok);
    __syncthreads();

    // bitonic ascending sort of v[0..2047]
    for (int k = 2; k <= Sdim; k <<= 1) {
        for (int j = k >> 1; j > 0; j >>= 1) {
            for (int i = t; i < Sdim; i += 256) {
                int ixj = i ^ j;
                if (ixj > i) {
                    bool up = ((i & k) == 0);
                    float a = v[i], c = v[ixj];
                    if (up ? (a > c) : (a < c)) { v[i] = c; v[ixj] = a; }
                }
            }
            __syncthreads();
        }
    }
    const float thresh = v[Sdim - kper];   // kper in [204,409] always

    float Eacc = 0.0f, Aacc = 0.0f;
    for (int i = t; i < Sdim; i += 256) {
        float raw = ts_raw[(size_t)b * Sdim + i];
        float tn = raw / T;
        float ex = (raw >= thresh) ? 1.0f : 0.0f;
        ts_out[(size_t)b * Sdim + i] = tn;
        expl[(size_t)b * Sdim + i] = ex;
        Eacc += ex * tn;
        Aacc += (1.0f - ex) * tn;
    }
    __syncthreads();
    sred[t] = Eacc; __syncthreads();
    for (int o = 128; o > 0; o >>= 1) { if (t < o) sred[t] += sred[t + o]; __syncthreads(); }
    const float E = sred[0];
    __syncthreads();
    sred[t] = Aacc; __syncthreads();
    for (int o = 128; o > 0; o >>= 1) { if (t < o) sred[t] += sred[t + o]; __syncthreads(); }
    if (t == 0) { stats[2 * b] = E; stats[2 * b + 1] = sred[0]; }
}

// ---------------- kernel 4: fused 3-pool partial sums ----------------------
__global__ __launch_bounds__(256)
void pool_partial_kernel(const float* __restrict__ vec, const float* __restrict__ ts_n,
                         const float* __restrict__ expl, const float* __restrict__ stats,
                         float* __restrict__ partial) {
    const int sc = blockIdx.x, hc = blockIdx.y, b = blockIdx.z;
    const int t = threadIdx.x;
    const int h = hc * 256 + t;
    const int s0 = sc * 256;

    __shared__ float wE[256], wT[256], wA[256];
    {
        float tn = ts_n[(size_t)b * Sdim + s0 + t];
        float ex = expl[(size_t)b * Sdim + s0 + t];
        float E = stats[2 * b], A = stats[2 * b + 1];
        wT[t] = tn;
        wE[t] = ex * tn / E;
        wA[t] = (1.0f - ex) * tn / A;
    }
    __syncthreads();

    float aE = 0.0f, aT = 0.0f, aA = 0.0f;
    const float* vp = vec + ((size_t)b * Sdim + s0) * Hdim + h;
#pragma unroll 4
    for (int s = 0; s < 256; ++s) {
        float x = vp[(size_t)s * Hdim];
        aE += wE[s] * x; aT += wT[s] * x; aA += wA[s] * x;
    }
    size_t base = (((size_t)b * 8 + sc) * 3) * Hdim + h;
    partial[base + 0 * Hdim] = aE;
    partial[base + 1 * Hdim] = aT;
    partial[base + 2 * Hdim] = aA;
}

// ---------------- kernel 5: reduce partials into d_out pools ----------------
__global__ __launch_bounds__(256)
void pool_reduce_kernel(const float* __restrict__ partial, float* __restrict__ d_out) {
    const int hc = blockIdx.x, pool = blockIdx.y, b = blockIdx.z;
    const int h = hc * 256 + threadIdx.x;
    float acc = 0.0f;
#pragma unroll
    for (int sc = 0; sc < 8; ++sc)
        acc += partial[(((size_t)b * 8 + sc) * 3 + pool) * Hdim + h];
    d_out[(size_t)Bdim * Sdim + (size_t)pool * Bdim * Hdim + (size_t)b * Hdim + h] = acc;
}

// ---------------- launcher ----------------
extern "C" void kernel_launch(void* const* d_in, const int* in_sizes, int n_in,
                              void* d_out, int out_size, void* d_ws, size_t ws_size,
                              hipStream_t stream) {
    const float* vec  = (const float*)d_in[0];
    const float* Wq   = (const float*)d_in[1];
    const float* bq   = (const float*)d_in[2];
    const float* Wk   = (const float*)d_in[3];
    const float* bk   = (const float*)d_in[4];
    const int*   mask = (const int*)d_in[5];
    float* out = (float*)d_out;

    char* ws = (char*)d_ws;
    size_t off = 0;
    const size_t qk_bytes = (size_t)Bdim * Sdim * Dpad * sizeof(_Float16); // 4 MB
    _Float16* qf     = (_Float16*)(ws + off); off += qk_bytes;
    _Float16* kf     = (_Float16*)(ws + off); off += qk_bytes;
    float*    ts_raw = (float*)(ws + off);    off += (size_t)Bdim * Sdim * 4;
    float*    expl   = (float*)(ws + off);    off += (size_t)Bdim * Sdim * 4;
    float*    stats  = (float*)(ws + off);    off += 1024;
    float*    part   = (float*)(ws + off);    off += (size_t)Bdim * 8 * 3 * Hdim * 4;
    _Float16* Wt     = (_Float16*)(ws + off); off += (size_t)128 * Hdim * sizeof(_Float16);

    // 0) transpose/pad weights to f16 [128][768]
    wt_kernel<<<128, 256, 0, stream>>>(Wq, Wk, Wt);

    // 1) WMMA Q/K projection (f16 outputs padded to 64)
    proj_kernel<<<(Bdim * Sdim) / 16, 256, 0, stream>>>(vec, Wt, bq, bk, qf, kf);

    // 2) WMMA scores + online softmax + dropout row sums -> ts_raw
    score_kernel<<<dim3(Sdim / 16, Bdim), 256, 0, stream>>>(qf, kf, mask, ts_raw);

    // 3) per-batch sort / threshold / normalize -> ts (d_out), expl, stats
    select_kernel<<<Bdim, 256, 0, stream>>>(ts_raw, mask, out, expl, stats);

    // 4) fused three-pool partial sums over vec_seq
    pool_partial_kernel<<<dim3(8, 3, Bdim), 256, 0, stream>>>(vec, out, expl, stats, part);

    // 5) reduce partials into exp/all/anti pools
    pool_reduce_kernel<<<dim3(3, 3, Bdim), 256, 0, stream>>>(part, out);
}
